// HQA_60181081752143
// MI455X (gfx1250) — matrix operand order repaired and checked by reading, not compiled, not run on hardware
//
#include <hip/hip_runtime.h>
#include <math.h>

// ---------------------------------------------------------------------------
// HQA forward for MI455X (gfx1250, wave32, WMMA + TDM).
// Outputs (flat, in order): rec[16,3,256,256], z_e_lower[16,3,256,256],
//   z_q[16,64,128,128], z_e[16,64,128,128], hard_idx[16,128,128], kl, commit
// ---------------------------------------------------------------------------

typedef __attribute__((ext_vector_type(16))) _Float16 v16h;
typedef __attribute__((ext_vector_type(8)))  float    v8f;
typedef __attribute__((ext_vector_type(4)))  unsigned u32x4;
typedef __attribute__((ext_vector_type(8)))  int      i32x8;
typedef __attribute__((ext_vector_type(4)))  int      i32x4;

#define TEMP_INV 1.4992503748125937f   // 1 / 0.667
#define LOG_K    5.545177444479562f    // ln(256)

__device__ __forceinline__ float mishf(float x) {
  float sp = (x > 20.0f) ? x : log1pf(__expf(x));
  return x * tanhf(sp);
}
__device__ __forceinline__ float sigm(float x) {
  return 1.0f / (1.0f + __expf(-x));
}

// ------------------------- encoder convs (VALU) ----------------------------

// conv 3->8, 3x3, stride 2, pad 1, mish. out [16,8,128,128]
__global__ void enc_conv1(const float* __restrict__ img, const float* __restrict__ w,
                          const float* __restrict__ bia, float* __restrict__ out) {
  int idx = blockIdx.x * blockDim.x + threadIdx.x;
  int x = idx & 127, y = (idx >> 7) & 127, o = (idx >> 14) & 7, b = idx >> 17;
  float s = bia[o];
  #pragma unroll
  for (int c = 0; c < 3; ++c)
    #pragma unroll
    for (int ky = 0; ky < 3; ++ky) {
      int iy = 2 * y + ky - 1;
      if (iy < 0 || iy > 255) continue;
      #pragma unroll
      for (int kx = 0; kx < 3; ++kx) {
        int ix = 2 * x + kx - 1;
        if (ix < 0 || ix > 255) continue;
        s += img[((b * 3 + c) << 16) + (iy << 8) + ix] * w[(o * 3 + c) * 9 + ky * 3 + kx];
      }
    }
  out[idx] = mishf(s);
}

// conv 8->16, 3x3, pad 1, mish. out [16,16,128,128]
__global__ void enc_conv2(const float* __restrict__ in, const float* __restrict__ w,
                          const float* __restrict__ bia, float* __restrict__ out) {
  int idx = blockIdx.x * blockDim.x + threadIdx.x;
  int x = idx & 127, y = (idx >> 7) & 127, o = (idx >> 14) & 15, b = idx >> 18;
  float s = bia[o];
  for (int c = 0; c < 8; ++c)
    #pragma unroll
    for (int ky = 0; ky < 3; ++ky) {
      int iy = y + ky - 1;
      if (iy < 0 || iy > 127) continue;
      #pragma unroll
      for (int kx = 0; kx < 3; ++kx) {
        int ix = x + kx - 1;
        if (ix < 0 || ix > 127) continue;
        s += in[((b * 8 + c) << 14) + (iy << 7) + ix] * w[(o * 8 + c) * 9 + ky * 3 + kx];
      }
    }
  out[idx] = mishf(s);
}

// conv 16->64, 1x1 (no pad), linear. out = z_e [16,64,128,128]
__global__ void enc_conv3(const float* __restrict__ in, const float* __restrict__ w,
                          const float* __restrict__ bia, float* __restrict__ ze) {
  int idx = blockIdx.x * blockDim.x + threadIdx.x;
  int p = idx & 16383, o = (idx >> 14) & 63, b = idx >> 20;
  float s = bia[o];
  #pragma unroll
  for (int c = 0; c < 16; ++c)
    s += in[((b * 16 + c) << 14) + p] * w[o * 16 + c];
  ze[idx] = s;
}

// codebook row norms + f16 codebook staging + zero the kl/commit accumulators
__global__ void cb_norms(const float* __restrict__ cb, float* __restrict__ cn,
                         float* __restrict__ acc, _Float16* __restrict__ cb16) {
  int k = threadIdx.x;
  float s = 0.0f;
  #pragma unroll 8
  for (int d = 0; d < 64; ++d) {
    float v = cb[k * 64 + d];
    s += v * v;
    cb16[k * 64 + d] = (_Float16)v;
  }
  cn[k] = s;
  if (k < 2) acc[k] = 0.0f;
}

// ------------------------------ VQ core ------------------------------------
// One wave32 per 16 pixel-rows. Codebook staged into LDS by the Tensor Data
// Mover (32 KB, one 1-D tile of 4096 8-byte elements). Two f16 WMMA GEMMs:
//   S = z(16x64) @ codebookT(64x256)   -> dists in VGPRs
//   z_q = soft(16x256) @ codebook(256x64)
__global__ __launch_bounds__(32, 1) void vq_kernel(
    const float* __restrict__ zE,        // [16,64,128,128]
    const float* __restrict__ u,         // [N,256]
    const _Float16* __restrict__ cb16,   // [256,64] f16 (workspace)
    const float* __restrict__ cnorm,     // [256]
    float* __restrict__ zq,              // [16,64,128,128]
    float* __restrict__ hidx,            // [N]
    float* __restrict__ acc)             // acc[0]=kl sum, acc[1]=commit sum
{
  __shared__ __align__(16) _Float16 s_cb[256][64];   // codebook, f16
  __shared__ float    s_cn[256];
  __shared__ float    s_zn[16];
  __shared__ float    s_gl[16][256];   // gumbel logits, then exp()
  __shared__ _Float16 s_soft[16][256]; // soft one-hot, f16

  const int lane = threadIdx.x;
  const int l15  = lane & 15;
  const int half = lane >> 4;
  const int rowbase = blockIdx.x * 16;

#if __has_builtin(__builtin_amdgcn_tensor_load_to_lds)
  {
    // Tensor DMA descriptor (ISA 08_async_tensor 8.3/8.4): 1-D tile, 8B elems.
    unsigned lds_off = (unsigned)(size_t)(&s_cb[0][0]); // low 32b of flat = LDS addr
    unsigned long long ga = (unsigned long long)(size_t)cb16;
    u32x4 g0;
    g0[0] = 1u;                                          // count=1, user mode
    g0[1] = lds_off;                                     // lds_addr
    g0[2] = (unsigned)(ga & 0xFFFFFFFFu);                // global_addr[31:0]
    g0[3] = (unsigned)((ga >> 32) & 0x01FFFFFFu) | 0x80000000u; // ga[56:32] | type=2
    i32x8 g1;
    g1[0] = (int)(3u << 16);      // workgroup_mask=0, data_size=3 (8B)
    g1[1] = (int)(4096u << 16);   // tensor_dim0[15:0]=4096 (8B units)
    g1[2] = (int)(1u << 16);      // tensor_dim0 hi=0 | tensor_dim1[15:0]=1
    g1[3] = (int)(4096u << 16);   // tensor_dim1 hi=0 | tile_dim0=4096
    g1[4] = 0;                    // tile_dim1=0 (unused), tile_dim2=0
    g1[5] = 4096;                 // tensor_dim0_stride lo
    g1[6] = 0;
    g1[7] = 0;
    i32x4 g2 = {};                // dims 2..4 unused
    i32x4 g3 = {};
    i32x8 g4 = {};                // clang-23 6-arg form: extra group, zero
    __builtin_amdgcn_tensor_load_to_lds(g0, g1, g2, g3, g4, 0);
  }
#else
  for (int i = lane; i < 256 * 64; i += 32)
    (&s_cb[0][0])[i] = cb16[i];
#endif

  for (int i = lane; i < 256; i += 32)
    s_cn[i] = cnorm[i];
  if (lane < 16) {
    int n = rowbase + lane;
    int b = n >> 14, hw = n & 16383;
    const float* zp = zE + ((size_t)b << 20) + hw;
    float s = 0.0f;
    for (int d = 0; d < 64; ++d) { float v = zp[d << 14]; s += v * v; }
    s_zn[lane] = s;
  }
#if __has_builtin(__builtin_amdgcn_tensor_load_to_lds)
  __builtin_amdgcn_s_wait_tensorcnt(0);
#endif
  __syncthreads();

  // ---- A fragments for GEMM1: z tile 16x64 (two 16x32 chunks) ----
  // A layout: lane holds row M=l15; VGPR i pair: K = (i<4 ? 2i : 16+2(i-4)) + 8*half
  v16h afrag[2];
  {
    int n = rowbase + l15;
    int b = n >> 14, hw = n & 16383;
    const float* zp = zE + ((size_t)b << 20) + hw;
    #pragma unroll
    for (int kc = 0; kc < 2; ++kc)
      #pragma unroll
      for (int i = 0; i < 8; ++i) {
        int k0 = kc * 32 + ((i < 4) ? 2 * i : 16 + 2 * (i - 4)) + 8 * half;
        afrag[kc][2 * i]     = (_Float16)zp[k0 << 14];
        afrag[kc][2 * i + 1] = (_Float16)zp[(k0 + 1) << 14];
      }
  }

  // ---- GEMM1: dists[m][k] = ||c_k||^2 + ||z_m||^2 - 2 z.c  (16 n-tiles) ----
  float dist[16][8];
  #pragma unroll
  for (int nt = 0; nt < 16; ++nt) {
    v8f c = {};
    #pragma unroll
    for (int kc = 0; kc < 2; ++kc) {
      // B layout: lane holds col N=l15; VGPR j pair: K = (j<4 ? 2j : 16+2(j-4)) + 8*half
      v16h bfrag;
      int slot = nt * 16 + l15;
      #pragma unroll
      for (int j = 0; j < 8; ++j) {
        int k = kc * 32 + ((j < 4) ? 2 * j : 16 + 2 * (j - 4)) + 8 * half;
        bfrag[2 * j]     = s_cb[slot][k];
        bfrag[2 * j + 1] = s_cb[slot][k + 1];
      }
      c = __builtin_amdgcn_wmma_f32_16x16x32_f16(false, afrag[kc], false, bfrag,
                                                 (short)0, c, false, false);
    }
    float cn = s_cn[nt * 16 + l15];
    #pragma unroll
    for (int r = 0; r < 8; ++r)               // D layout: M = r + 8*half, N = l15
      dist[nt][r] = cn + s_zn[r + 8 * half] - 2.0f * c[r];
  }

  // ---- plain softmax (probs) stats + kl & commit partial sums ----
  float rmax[8], rsum[8];
  #pragma unroll
  for (int r = 0; r < 8; ++r) {
    float m = -dist[0][r];
    #pragma unroll
    for (int nt = 1; nt < 16; ++nt) m = fmaxf(m, -dist[nt][r]);
    #pragma unroll
    for (int s = 1; s < 16; s <<= 1) m = fmaxf(m, __shfl_xor(m, s));
    float se = 0.0f;
    #pragma unroll
    for (int nt = 0; nt < 16; ++nt) se += __expf(-dist[nt][r] - m);
    #pragma unroll
    for (int s = 1; s < 16; s <<= 1) se += __shfl_xor(se, s);
    rmax[r] = m; rsum[r] = se;
  }
  float klp = 0.0f, cmp = 0.0f;
  #pragma unroll
  for (int r = 0; r < 8; ++r) {
    float inv = 1.0f / rsum[r];
    #pragma unroll
    for (int nt = 0; nt < 16; ++nt) {
      float p = __expf(-dist[nt][r] - rmax[r]) * inv;
      klp += p * (logf(p + 1e-9f) + LOG_K);
      cmp += p * dist[nt][r];
    }
  }
  #pragma unroll
  for (int s = 1; s < 32; s <<= 1) { klp += __shfl_xor(klp, s); cmp += __shfl_xor(cmp, s); }
  if (lane == 0) { atomicAdd(&acc[0], klp); atomicAdd(&acc[1], cmp); }

  // ---- gumbel logits -> LDS; local argmax/max ----
  float bmax[8]; int bidx[8];
  #pragma unroll
  for (int r = 0; r < 8; ++r) { bmax[r] = -1e30f; bidx[r] = 0; }
  {
    const float* up = u + (size_t)rowbase * 256;
    #pragma unroll
    for (int r = 0; r < 8; ++r) {
      int m = r + 8 * half;
      #pragma unroll
      for (int nt = 0; nt < 16; ++nt) {
        int k = nt * 16 + l15;
        float uu = up[m * 256 + k];
        float g  = -__logf(-__logf(uu));
        float gl = (g - dist[nt][r]) * TEMP_INV;
        s_gl[m][k] = gl;
        if (gl > bmax[r]) { bmax[r] = gl; bidx[r] = k; }
      }
    }
  }
  // row-wise (val,idx) reduce within 16-lane halves; ties -> lower index
  #pragma unroll
  for (int r = 0; r < 8; ++r) {
    float v = bmax[r]; int ix = bidx[r];
    #pragma unroll
    for (int s = 1; s < 16; s <<= 1) {
      float ov = __shfl_xor(v, s);
      int   oi = __shfl_xor(ix, s);
      if (ov > v || (ov == v && oi < ix)) { v = ov; ix = oi; }
    }
    bmax[r] = v; bidx[r] = ix;
  }
  if (l15 == 0) {
    #pragma unroll
    for (int r = 0; r < 8; ++r)
      hidx[rowbase + r + 8 * half] = (float)bidx[r];
  }
  __syncthreads();

  // ---- gumbel softmax: exp pass + row sums, then normalize to f16 ----
  float gsum[8];
  #pragma unroll
  for (int r = 0; r < 8; ++r) {
    int m = r + 8 * half;
    float se = 0.0f;
    #pragma unroll
    for (int nt = 0; nt < 16; ++nt) {
      int k = nt * 16 + l15;
      float e = __expf(s_gl[m][k] - bmax[r]);
      s_gl[m][k] = e;
      se += e;
    }
    #pragma unroll
    for (int s = 1; s < 16; s <<= 1) se += __shfl_xor(se, s);
    gsum[r] = se;
  }
  __syncthreads();
  #pragma unroll
  for (int r = 0; r < 8; ++r) {
    int m = r + 8 * half;
    float inv = 1.0f / gsum[r];
    #pragma unroll
    for (int nt = 0; nt < 16; ++nt) {
      int k = nt * 16 + l15;
      s_soft[m][k] = (_Float16)(s_gl[m][k] * inv);
    }
  }
  __syncthreads();

  // ---- GEMM2: z_q(16x64) = soft(16x256) @ codebook(256x64) ----
  v8f c2[4] = {};
  #pragma unroll
  for (int kc = 0; kc < 8; ++kc) {
    v16h a2;
    #pragma unroll
    for (int i = 0; i < 8; ++i) {
      int k0 = kc * 32 + ((i < 4) ? 2 * i : 16 + 2 * (i - 4)) + 8 * half;
      a2[2 * i]     = s_soft[l15][k0];
      a2[2 * i + 1] = s_soft[l15][k0 + 1];
    }
    #pragma unroll
    for (int dt = 0; dt < 4; ++dt) {
      v16h b2;
      #pragma unroll
      for (int j = 0; j < 8; ++j) {
        int k = kc * 32 + ((j < 4) ? 2 * j : 16 + 2 * (j - 4)) + 8 * half;
        int d = dt * 16 + l15;
        b2[2 * j]     = s_cb[k][d];
        b2[2 * j + 1] = s_cb[k + 1][d];
      }
      c2[dt] = __builtin_amdgcn_wmma_f32_16x16x32_f16(false, a2, false, b2,
                                                      (short)0, c2[dt], false, false);
    }
  }
  // scatter-store z_q back to NCHW
  #pragma unroll
  for (int dt = 0; dt < 4; ++dt)
    #pragma unroll
    for (int r = 0; r < 8; ++r) {
      int m = r + 8 * half;
      int n = rowbase + m;
      int b = n >> 14, hw = n & 16383;
      int d = dt * 16 + l15;
      zq[((b * 64 + d) << 14) + hw] = c2[dt][r];
    }
}

__global__ void vq_finalize(const float* __restrict__ acc,
                            float* __restrict__ kl, float* __restrict__ cm) {
  if (threadIdx.x == 0) {
    kl[0] = acc[0] * (1.0f / 16.0f);
    cm[0] = acc[1] * (1.0f / 16.0f);
  }
}

// ---------------------- decoder convs (implicit-GEMM WMMA) -----------------

// conv 64->32, 3x3, pad 1, mish, as implicit GEMM:
//   M = 16 pixels/wave (one row segment), N = 32 out-ch (2 tiles), K = 64*9.
__global__ __launch_bounds__(128, 1) void dec_conv1_wmma(
    const float* __restrict__ in,   // z_q [16,64,128,128]
    const float* __restrict__ w,    // [32,64,3,3]
    const float* __restrict__ bia,  // [32]
    float* __restrict__ out)        // [16,32,128,128]
{
  __shared__ _Float16 s_w[9][64][32];  // [tap][cin][cout], 36 KB
  const int tid  = threadIdx.x;
  const int lane = tid & 31;
  const int wave = tid >> 5;
  const int l15  = lane & 15;
  const int half = lane >> 4;

  for (int i = tid; i < 9 * 64 * 32; i += 128) {
    int tap = i >> 11, c = (i >> 5) & 63, n = i & 31;
    s_w[tap][c][n] = (_Float16)w[(n * 64 + c) * 9 + tap];
  }
  __syncthreads();

  int base = (blockIdx.x * 4 + wave) * 16;   // pixel base over [16,128,128]
  int b  = base >> 14;
  int rem = base & 16383;
  int y  = rem >> 7;
  int xb = rem & 127;
  const float* inb = in + ((size_t)(b * 64) << 14);

  v8f acc[2] = {};
  for (int tap = 0; tap < 9; ++tap) {
    int ky = tap / 3, kx = tap % 3;
    int iy = y + ky - 1;
    bool yok = (iy >= 0 && iy < 128);
    int ix = xb + l15 + kx - 1;                 // A row M = l15 -> pixel x
    bool ok = yok && (ix >= 0 && ix < 128);
    int off = (iy << 7) + ix;
    #pragma unroll
    for (int ch = 0; ch < 2; ++ch) {            // two 32-channel K chunks
      v16h a;
      #pragma unroll
      for (int i = 0; i < 8; ++i) {
        int kk = ((i < 4) ? 2 * i : 16 + 2 * (i - 4)) + 8 * half;
        int c0 = ch * 32 + kk;
        a[2 * i]     = ok ? (_Float16)inb[(c0 << 14) + off] : (_Float16)0.0f;
        a[2 * i + 1] = ok ? (_Float16)inb[((c0 + 1) << 14) + off] : (_Float16)0.0f;
      }
      #pragma unroll
      for (int nt = 0; nt < 2; ++nt) {
        v16h bf;
        #pragma unroll
        for (int j = 0; j < 8; ++j) {
          int c = ch * 32 + ((j < 4) ? 2 * j : 16 + 2 * (j - 4)) + 8 * half;
          bf[2 * j]     = s_w[tap][c][nt * 16 + l15];
          bf[2 * j + 1] = s_w[tap][c + 1][nt * 16 + l15];
        }
        acc[nt] = __builtin_amdgcn_wmma_f32_16x16x32_f16(false, a, false, bf,
                                                         (short)0, acc[nt], false, false);
      }
    }
  }
  #pragma unroll
  for (int nt = 0; nt < 2; ++nt) {
    int oc = nt * 16 + l15;
    float bb = bia[oc];
    #pragma unroll
    for (int r = 0; r < 8; ++r) {
      int m = r + 8 * half;
      out[((b * 32 + oc) << 14) + (y << 7) + xb + m] = mishf(acc[nt][r] + bb);
    }
  }
}

// nearest-upsample x2 fused into conv 32->16, 3x3, pad 1, mish, implicit GEMM:
//   M = 16 pixels/wave (256-wide rows), N = 16 out-ch, K = 32 per tap (9 taps).
__global__ __launch_bounds__(128, 1) void dec_conv2_wmma(
    const float* __restrict__ in,   // [16,32,128,128]
    const float* __restrict__ w,    // [16,32,3,3]
    const float* __restrict__ bia,  // [16]
    float* __restrict__ out)        // [16,16,256,256]
{
  __shared__ _Float16 s_w[9][32][16];  // 9 KB
  const int tid  = threadIdx.x;
  const int lane = tid & 31;
  const int wave = tid >> 5;
  const int l15  = lane & 15;
  const int half = lane >> 4;

  for (int i = tid; i < 9 * 32 * 16; i += 128) {
    int tap = i >> 9, c = (i >> 4) & 31, n = i & 15;
    s_w[tap][c][n] = (_Float16)w[(n * 32 + c) * 9 + tap];
  }
  __syncthreads();

  int base = (blockIdx.x * 4 + wave) * 16;   // pixel base over [16,256,256]
  int b  = base >> 16;
  int rem = base & 65535;
  int y  = rem >> 8;
  int xb = rem & 255;
  const float* inb = in + ((size_t)(b * 32) << 14);

  v8f acc = {};
  for (int tap = 0; tap < 9; ++tap) {
    int ky = tap / 3, kx = tap % 3;
    int yy = y + ky - 1;
    bool yok = (yy >= 0 && yy < 256);
    int iy = yy >> 1;                            // nearest upsample fold
    int xx = xb + l15 + kx - 1;
    bool ok = yok && (xx >= 0 && xx < 256);
    int ix = xx >> 1;
    int off = (iy << 7) + ix;
    v16h a;
    #pragma unroll
    for (int i = 0; i < 8; ++i) {
      int c0 = ((i < 4) ? 2 * i : 16 + 2 * (i - 4)) + 8 * half;
      a[2 * i]     = ok ? (_Float16)inb[(c0 << 14) + off] : (_Float16)0.0f;
      a[2 * i + 1] = ok ? (_Float16)inb[((c0 + 1) << 14) + off] : (_Float16)0.0f;
    }
    v16h bf;
    #pragma unroll
    for (int j = 0; j < 8; ++j) {
      int c = ((j < 4) ? 2 * j : 16 + 2 * (j - 4)) + 8 * half;
      bf[2 * j]     = s_w[tap][c][l15];
      bf[2 * j + 1] = s_w[tap][c + 1][l15];
    }
    acc = __builtin_amdgcn_wmma_f32_16x16x32_f16(false, a, false, bf,
                                                 (short)0, acc, false, false);
  }
  int oc = l15;
  float bb = bia[oc];
  #pragma unroll
  for (int r = 0; r < 8; ++r) {
    int m = r + 8 * half;
    out[((b * 16 + oc) << 16) + (y << 8) + xb + m] = mishf(acc[r] + bb);
  }
}

// conv 16->3, 3x3, pad 1, sigmoid. out [16,3,256,256]  (VALU; only 3 out-ch)
__global__ void dec_conv3(const float* __restrict__ in, const float* __restrict__ w,
                          const float* __restrict__ bia, float* __restrict__ out) {
  int idx = blockIdx.x * blockDim.x + threadIdx.x;
  int x = idx & 255, y = (idx >> 8) & 255;
  int t = idx >> 16;
  int o = t % 3, b = t / 3;
  float s = bia[o];
  for (int c = 0; c < 16; ++c)
    #pragma unroll
    for (int ky = 0; ky < 3; ++ky) {
      int iy = y + ky - 1;
      if (iy < 0 || iy > 255) continue;
      #pragma unroll
      for (int kx = 0; kx < 3; ++kx) {
        int ix = x + kx - 1;
        if (ix < 0 || ix > 255) continue;
        s += in[((b * 16 + c) << 16) + (iy << 8) + ix] * w[(o * 16 + c) * 9 + ky * 3 + kx];
      }
    }
  out[idx] = sigm(s);
}

// ------------------------------- launcher ----------------------------------

extern "C" void kernel_launch(void* const* d_in, const int* in_sizes, int n_in,
                              void* d_out, int out_size, void* d_ws, size_t ws_size,
                              hipStream_t stream) {
  (void)in_sizes; (void)n_in; (void)out_size; (void)ws_size;
  const float* img    = (const float*)d_in[0];
  const float* u      = (const float*)d_in[1];
  const float* enc_w1 = (const float*)d_in[2];
  const float* enc_b1 = (const float*)d_in[3];
  const float* enc_w2 = (const float*)d_in[4];
  const float* enc_b2 = (const float*)d_in[5];
  const float* enc_w3 = (const float*)d_in[6];
  const float* enc_b3 = (const float*)d_in[7];
  const float* cb     = (const float*)d_in[8];
  const float* dec_w1 = (const float*)d_in[9];
  const float* dec_b1 = (const float*)d_in[10];
  const float* dec_w2 = (const float*)d_in[11];
  const float* dec_b2 = (const float*)d_in[12];
  const float* dec_w3 = (const float*)d_in[13];
  const float* dec_b3 = (const float*)d_in[14];

  float* out  = (float*)d_out;
  float* rec  = out;               // 16*3*256*256   = 3,145,728
  float* zel  = out + 3145728;     // 3,145,728
  float* zq   = out + 6291456;     // 16*64*128*128  = 16,777,216
  float* ze   = out + 23068672;    // 16,777,216
  float* hidx = out + 39845888;    // 262,144
  float* klo  = out + 40108032;    // 1
  float* cmo  = out + 40108033;    // 1

  float* ws     = (float*)d_ws;
  float* t_enc1 = ws;                          // 2,097,152
  float* t_enc2 = ws + 2097152;                // 4,194,304
  float* t_dec1 = ws + 6291456;                // 8,388,608
  float* t_dec2 = ws + 14680064;               // 16,777,216
  float* t_cn   = ws + 31457280;               // 256
  float* t_acc  = ws + 31457536;               // 2
  _Float16* t_cb16 = (_Float16*)(ws + 31457544); // 16384 f16 (32 KB, 32B-aligned)

  (void)hipMemcpyAsync(zel, img, (size_t)3145728 * sizeof(float),
                       hipMemcpyDeviceToDevice, stream);

  enc_conv1<<<8192, 256, 0, stream>>>(img, enc_w1, enc_b1, t_enc1);
  enc_conv2<<<16384, 256, 0, stream>>>(t_enc1, enc_w2, enc_b2, t_enc2);
  enc_conv3<<<65536, 256, 0, stream>>>(t_enc2, enc_w3, enc_b3, ze);

  cb_norms<<<1, 256, 0, stream>>>(cb, t_cn, t_acc, t_cb16);
  vq_kernel<<<16384, 32, 0, stream>>>(ze, u, t_cb16, t_cn, zq, hidx, t_acc);
  vq_finalize<<<1, 32, 0, stream>>>(t_acc, klo, cmo);

  dec_conv1_wmma<<<4096, 128, 0, stream>>>(zq, dec_w1, dec_b1, t_dec1);
  dec_conv2_wmma<<<16384, 128, 0, stream>>>(t_dec1, dec_w2, dec_b2, t_dec2);
  dec_conv3<<<12288, 256, 0, stream>>>(t_dec2, dec_w3, dec_b3, rec);
}
